// NetworkGRU__58652073394484
// MI455X (gfx1250) — matrix operand order repaired and checked by reading, not compiled, not run on hardware
//
#include <hip/hip_runtime.h>
#include <hip/hip_bf16.h>

// ---------------------------------------------------------------------------
// RNN: h_t = sigmoid(x_t @ Wi + h_{t-1} @ Wh); out = (sigmoid(h@fc_w+b), h)
// Phase 1: parallel projection GEMM (131072x320x256, bf16 WMMA, f32 accum),
//          with async global->LDS staging of the Wi^T tiles (ASYNCcnt path).
// Phase 2: 512-step recurrence on one WGP, Wh^T + h resident in 320KB LDS,
//          prefetching the next timestep's Xpre slice under the WMMA work.
// ---------------------------------------------------------------------------

typedef __bf16 bf16_t;
typedef __attribute__((ext_vector_type(8)))  __bf16 v8bf;
typedef __attribute__((ext_vector_type(16))) __bf16 v16bf;
typedef __attribute__((ext_vector_type(8)))  float  v8f;

#define N_EMBED   300
#define KPAD      320
#define N_HIDDEN  256
#define BATCH     256
#define SEQ       512
#define MTILE     64          // rows per block in projection GEMM

__device__ __forceinline__ v16bf make_frag(v8bf lo, v8bf hi) {
  union { v16bf v; v8bf h[2]; } u;
  u.h[0] = lo; u.h[1] = hi;
  return u.v;
}

__device__ __forceinline__ float sigmoidf_(float x) {
  return 1.0f / (1.0f + __expf(-x));
}

// Async DMA: 16 bytes global -> LDS, tracked by ASYNCcnt (CDNA5 ISA).
// lds_off is the wave-relative LDS byte offset (low 32 bits of a generic
// pointer into __shared__, per the LDS aperture rules in CDNA5 ISA).
__device__ __forceinline__ void async_b128_to_lds(unsigned lds_off, const void* gsrc) {
  asm volatile("global_load_async_to_lds_b128 %0, %1, off"
               :: "v"(lds_off), "v"(gsrc)
               : "memory");
}
__device__ __forceinline__ void wait_asynccnt0() {
  asm volatile("s_wait_asynccnt 0x0" ::: "memory");
}

// ---------------------------------------------------------------------------
// Prep: Wi (f32, [300][256]) -> WiT bf16 [256][KPAD] (transposed, zero-padded)
// ---------------------------------------------------------------------------
__global__ __launch_bounds__(KPAD) void prep_wi_kernel(
    const float* __restrict__ Wi, bf16_t* __restrict__ WiT)
{
  const int n = blockIdx.x;      // 0..255 (hidden unit)
  const int k = threadIdx.x;     // 0..319 (embed dim, padded)
  WiT[n * KPAD + k] = (k < N_EMBED) ? (bf16_t)Wi[k * N_HIDDEN + n] : (bf16_t)0.0f;
}

// ---------------------------------------------------------------------------
// Projection: Xpre[s][b][n] = sum_e emb[words[b][s]][e] * Wi[e][n]
// M = s*256+b.  Block: 64 rows x 256 cols, 256 threads (8 waves).
// Wave w: Mtile = w>>1 (16 rows), N half = (w&1)*128 (8 N-tiles).
// ---------------------------------------------------------------------------
__global__ __launch_bounds__(256) void proj_kernel(
    const int*   __restrict__ words,   // [BATCH][SEQ]
    const float* __restrict__ emb,     // [V][N_EMBED]
    const bf16_t* __restrict__ WiT,    // [256][KPAD]
    float* __restrict__ Xpre)          // [SEQ][BATCH][N_HIDDEN]
{
  __shared__ __attribute__((aligned(16))) int    rowWord[MTILE];
  __shared__ __attribute__((aligned(16))) bf16_t Asub[MTILE][40];      // 32 + pad
  __shared__ __attribute__((aligned(16))) bf16_t WiTs[N_HIDDEN][40];   // 32 + pad

  const int t    = threadIdx.x;
  const int w    = t >> 5;
  const int lane = t & 31;
  const int half = lane >> 4;
  const int ln   = lane & 15;
  const int m0   = blockIdx.x * MTILE;

  if (t < MTILE) {
    const int m = m0 + t;
    const int b = m & (BATCH - 1);
    const int s = m >> 8;
    rowWord[t] = words[b * SEQ + s];
  }
  __syncthreads();

  const int mt = w >> 1;          // 0..3
  const int nh = w & 1;           // 0/1 -> cols nh*128

  v8f acc[8];
  const v8f vzero = {0.f, 0.f, 0.f, 0.f, 0.f, 0.f, 0.f, 0.f};
#pragma unroll
  for (int i = 0; i < 8; ++i) acc[i] = vzero;

  for (int kc = 0; kc < KPAD / 32; ++kc) {
    const int k0 = kc * 32;

    // Async-stage WiT chunk (bf16 raw copy): row t, 32 contiguous bf16 as
    // 4 x global_load_async_to_lds_b128 — no VGPR round-trip, ASYNCcnt path.
    {
      const char* g = (const char*)(WiT + (long)t * KPAD + k0);
      const unsigned l = (unsigned)(unsigned long long)&WiTs[t][0];
#pragma unroll
      for (int i = 0; i < 4; ++i)
        async_b128_to_lds(l + i * 16, g + i * 16);
    }

    // Stage gathered embedding tile (64x32) as bf16, zero-padded past K=300
    // (needs f32->bf16 conversion, so this one stays through VGPRs).
    {
      const int row = t >> 2;
      const int c0  = (t & 3) * 8;
      const long word = rowWord[row];
      const float* src = emb + word * (long)N_EMBED;
#pragma unroll
      for (int j = 0; j < 8; ++j) {
        const int e = k0 + c0 + j;
        const float v = (e < N_EMBED) ? src[e] : 0.0f;
        Asub[row][c0 + j] = (bf16_t)v;
      }
    }

    wait_asynccnt0();   // async LDS writes landed before the barrier
    __syncthreads();

    // A fragment (16x32 bf16, ISA 7.12.2 layout): 2 x 16B LDS loads per lane.
    const v8bf alo = *(const v8bf*)&Asub[mt * 16 + ln][half * 8];
    const v8bf ahi = *(const v8bf*)&Asub[mt * 16 + ln][16 + half * 8];
    const v16bf a = make_frag(alo, ahi);
#pragma unroll
    for (int i = 0; i < 8; ++i) {
      const int n = nh * 128 + i * 16 + ln;
      const v8bf blo = *(const v8bf*)&WiTs[n][half * 8];
      const v8bf bhi = *(const v8bf*)&WiTs[n][16 + half * 8];
      const v16bf bb = make_frag(blo, bhi);
      acc[i] = __builtin_amdgcn_wmma_f32_16x16x32_bf16(
          false, a, false, bb, (short)0, acc[i], false, false);
    }
    __syncthreads();
  }

  // Store f32 result tiles: C/D layout M = r + 8*half, N = lane&15.
#pragma unroll
  for (int i = 0; i < 8; ++i) {
    const int n = nh * 128 + i * 16 + ln;
#pragma unroll
    for (int r = 0; r < 8; ++r) {
      const long m = m0 + mt * 16 + r + 8 * half;
      Xpre[m * N_HIDDEN + n] = acc[i][r];
    }
  }
}

// ---------------------------------------------------------------------------
// Recurrence: one workgroup, 512 threads (16 waves). WhT + h in LDS (bf16).
// Wave w owns rows [16w,16w+16) x all 256 cols = 16 accumulator tiles.
// ---------------------------------------------------------------------------
__global__ __launch_bounds__(512) void recur_kernel(
    const float* __restrict__ Xpre,    // [SEQ][256][256]
    const float* __restrict__ Wh,      // [256][256]
    const float* __restrict__ fc_w,    // [256]
    const float* __restrict__ fc_b,    // [1]
    float* __restrict__ out)           // [256] sig, then [256*256] hidden
{
  __shared__ __attribute__((aligned(16))) bf16_t h_s[256][264];   // 132 KB
  __shared__ __attribute__((aligned(16))) bf16_t WhT[256][264];   // 132 KB
  __shared__ float partial[512];

  const int t    = threadIdx.x;
  const int w    = t >> 5;
  const int lane = t & 31;
  const int half = lane >> 4;
  const int ln   = lane & 15;
  const int mt   = w;               // 0..15 (16-row strip)

  // Load Wh transposed into LDS as bf16; zero h.
  for (int idx = t; idx < 256 * 256; idx += 512) {
    const int n = idx >> 8;
    const int k = idx & 255;
    WhT[n][k] = (bf16_t)Wh[k * 256 + n];
    h_s[n][k] = (bf16_t)0.0f;
  }
  __syncthreads();

  for (int s = 0; s < SEQ; ++s) {
    const float* xp = Xpre + (long)s * 256 * 256;

    // Seed accumulators with Xpre[s] (input projection) — WMMA adds h@Wh.
    v8f acc[16];
#pragma unroll
    for (int nt = 0; nt < 16; ++nt) {
#pragma unroll
      for (int r = 0; r < 8; ++r)
        acc[nt][r] = xp[(mt * 16 + r + 8 * half) * 256 + nt * 16 + ln];
    }

    // Prefetch next step's 256KB Xpre slice under this step's WMMA work:
    // 512 threads x 4 x 128B cachelines = 256KB (global_prefetch_b8, untracked).
    if (s + 1 < SEQ) {
      const char* nxt = (const char*)(Xpre + (long)(s + 1) * 256 * 256);
#pragma unroll
      for (int pf = 0; pf < 4; ++pf)
        __builtin_prefetch(nxt + ((long)t * 512 + pf * 128), 0, 0);
    }

#pragma unroll
    for (int kc = 0; kc < 8; ++kc) {
      const int k0 = kc * 32;
      const v8bf alo = *(const v8bf*)&h_s[mt * 16 + ln][k0 + half * 8];
      const v8bf ahi = *(const v8bf*)&h_s[mt * 16 + ln][k0 + 16 + half * 8];
      const v16bf a = make_frag(alo, ahi);
#pragma unroll
      for (int nt = 0; nt < 16; ++nt) {
        const v8bf blo = *(const v8bf*)&WhT[nt * 16 + ln][k0 + half * 8];
        const v8bf bhi = *(const v8bf*)&WhT[nt * 16 + ln][k0 + 16 + half * 8];
        const v16bf bb = make_frag(blo, bhi);
        acc[nt] = __builtin_amdgcn_wmma_f32_16x16x32_bf16(
            false, a, false, bb, (short)0, acc[nt], false, false);
      }
    }

    __syncthreads();   // all waves finished reading h_s

#pragma unroll
    for (int nt = 0; nt < 16; ++nt) {
#pragma unroll
      for (int r = 0; r < 8; ++r) {
        const int m = mt * 16 + r + 8 * half;
        const int n = nt * 16 + ln;
        const float hv = sigmoidf_(acc[nt][r]);
        h_s[m][n] = (bf16_t)hv;
        if (s == SEQ - 1) out[256 + m * 256 + n] = hv;   // f32 hidden output
      }
    }
    __syncthreads();   // h_s update visible before next step
  }

  // FC head: sig[b] = sigmoid(hidden[b,:] . fc_w + fc_b). 2 threads per row.
  {
    const int b = t >> 1;
    const int hhalf = t & 1;
    float sum = 0.0f;
    const int kb = hhalf * 128;
    for (int k = kb; k < kb + 128; ++k)
      sum += (float)h_s[b][k] * fc_w[k];
    partial[t] = sum;
  }
  __syncthreads();
  if (t < 256) {
    const float z = partial[2 * t] + partial[2 * t + 1] + fc_b[0];
    out[t] = sigmoidf_(z);
  }
}

// ---------------------------------------------------------------------------
extern "C" void kernel_launch(void* const* d_in, const int* in_sizes, int n_in,
                              void* d_out, int out_size, void* d_ws, size_t ws_size,
                              hipStream_t stream) {
  const int*   words = (const int*)  d_in[0];
  const float* emb   = (const float*)d_in[1];
  const float* Wi    = (const float*)d_in[2];
  const float* Wh    = (const float*)d_in[3];
  const float* fc_w  = (const float*)d_in[4];
  const float* fc_b  = (const float*)d_in[5];
  float* out = (float*)d_out;

  // Workspace: [0, 256KB): WiT bf16 [256][320]; [256KB, +128MB): Xpre f32.
  char* ws = (char*)d_ws;
  bf16_t* WiT = (bf16_t*)ws;
  float*  Xpre = (float*)(ws + 256 * 1024);

  prep_wi_kernel<<<256, KPAD, 0, stream>>>(Wi, WiT);
  proj_kernel<<<(SEQ * BATCH) / MTILE, 256, 0, stream>>>(words, emb, WiT, Xpre);
  recur_kernel<<<1, 512, 0, stream>>>(Xpre, Wh, fc_w, fc_b, out);
}
